// GCN_37769942401381
// MI455X (gfx1250) — matrix operand order repaired and checked by reading, not compiled, not run on hardware
//
#include <hip/hip_runtime.h>

typedef __attribute__((ext_vector_type(16))) __bf16 v16bf;
typedef __attribute__((ext_vector_type(8)))  float  v8f;

static __device__ __forceinline__ __bf16 f2bf(float f) { return (__bf16)f; }

// ---------------- degree / norm ----------------

__global__ void k_deg_init(float* deg, int n) {
  int i = blockIdx.x * blockDim.x + threadIdx.x;
  if (i < n) deg[i] = 1.0f;   // self-loop contributes 1
}

__global__ void k_deg_accum(const long long* __restrict__ ei, float* deg, int E) {
  int e = blockIdx.x * blockDim.x + threadIdx.x;
  if (e < E) {
    int d = (int)ei[(size_t)E + e];
    unsafeAtomicAdd(&deg[d], 1.0f);
  }
}

__global__ void k_dinv(float* deg, int n) {
  int i = blockIdx.x * blockDim.x + threadIdx.x;
  if (i < n) deg[i] = rsqrtf(deg[i]);   // deg >= 1 always
}

// ---------------- GEMM1: xw1[N,128] = x[N,256] @ W1[256,128] (bf16 WMMA) ----------------
// 256 threads = 8 waves. Each wave owns a DISTINCT 16-row tile (block = 128 rows)
// and loops over all 8 column tiles -> x is read exactly once from HBM, and one
// A-fragment feeds 8 back-to-back v_wmma ops (B fragments from LDS-staged W1).

__global__ void __launch_bounds__(256)
k_gemm1(const float* __restrict__ x, const float* __restrict__ W1,
        float* __restrict__ xw1, int n) {
  __shared__ __bf16 w1t[128 * 128];   // [col][k-local], 32 KB

  const int t    = threadIdx.x;
  const int wave = t >> 5;
  const int lane = t & 31;
  const int lrow = lane & 15;
  const int half = lane >> 4;          // 0 or 1

  const int row0 = blockIdx.x * 128 + wave * 16;
  int arow = row0 + lrow; if (arow >= n) arow = n - 1;
  const float* xrow = x + (size_t)arow * 256;

  v8f acc[8];
#pragma unroll
  for (int ct = 0; ct < 8; ++ct) acc[ct] = v8f{};

  for (int kh = 0; kh < 2; ++kh) {
    // stage W1[kh*128 + kk, c] -> w1t[c*128 + kk] as bf16 (transposed)
    for (int i = t; i < 128 * 128; i += 256) {
      int c  = i >> 7;
      int kk = i & 127;
      w1t[c * 128 + kk] = f2bf(W1[(size_t)(kh * 128 + kk) * 128 + c]);
    }
    __syncthreads();

    for (int k0 = 0; k0 < 128; k0 += 32) {
      const int kg = kh * 128 + k0;
      // A fragment: elem s -> K = kg + half*16 + s
      v16bf a;
      const int ka = kg + half * 16;
#pragma unroll
      for (int j = 0; j < 8; ++j) {
        a[j]     = f2bf(xrow[ka + j]);
        a[8 + j] = f2bf(xrow[ka + 8 + j]);
      }
      // one A fragment x 8 column tiles
#pragma unroll
      for (int ct = 0; ct < 8; ++ct) {
        v16bf b;
        const __bf16* bp = &w1t[(ct * 16 + lrow) * 128 + k0 + half * 16];
#pragma unroll
        for (int i = 0; i < 16; ++i) b[i] = bp[i];
        acc[ct] = __builtin_amdgcn_wmma_f32_16x16x32_bf16(
            false, a, false, b, (short)0, acc[ct], false, false);
      }
    }
    __syncthreads();
  }

  // D store: lane -> N = lrow, VGPR r -> M = r + half*8
#pragma unroll
  for (int ct = 0; ct < 8; ++ct) {
#pragma unroll
    for (int r = 0; r < 8; ++r) {
      int m = row0 + r + half * 8;
      if (m < n) xw1[(size_t)m * 128 + ct * 16 + lrow] = acc[ct][r];
    }
  }
}

// ---------------- GEMM2: hw2[N,48] = h[N,128] @ W2pad[128,48] (bf16 WMMA) ----------------
// 256 threads = 8 waves, each wave a distinct 16-row tile, 3 column tiles
// (40 cols zero-padded to 48). h is read exactly once.

__global__ void __launch_bounds__(256)
k_gemm2(const float* __restrict__ h, const float* __restrict__ W2,
        float* __restrict__ hw2, int n) {
  __shared__ __bf16 w2t[48 * 128];    // [col][k], 12 KB

  const int t    = threadIdx.x;
  const int wave = t >> 5;
  const int lane = t & 31;
  const int lrow = lane & 15;
  const int half = lane >> 4;

  for (int i = t; i < 48 * 128; i += 256) {
    int c = i >> 7;
    int k = i & 127;
    w2t[c * 128 + k] = f2bf(c < 40 ? W2[(size_t)k * 40 + c] : 0.0f);
  }
  __syncthreads();

  const int row0 = blockIdx.x * 128 + wave * 16;
  int arow = row0 + lrow; if (arow >= n) arow = n - 1;
  const float* hrow = h + (size_t)arow * 128;

  v8f acc[3];
#pragma unroll
  for (int ct = 0; ct < 3; ++ct) acc[ct] = v8f{};

  for (int k0 = 0; k0 < 128; k0 += 32) {
    v16bf a;
    const int ka = k0 + half * 16;
#pragma unroll
    for (int j = 0; j < 8; ++j) {
      a[j]     = f2bf(hrow[ka + j]);
      a[8 + j] = f2bf(hrow[ka + 8 + j]);
    }
#pragma unroll
    for (int ct = 0; ct < 3; ++ct) {
      v16bf b;
      const __bf16* bp = &w2t[(ct * 16 + lrow) * 128 + k0 + half * 16];
#pragma unroll
      for (int i = 0; i < 16; ++i) b[i] = bp[i];
      acc[ct] = __builtin_amdgcn_wmma_f32_16x16x32_bf16(
          false, a, false, b, (short)0, acc[ct], false, false);
    }
  }

#pragma unroll
  for (int ct = 0; ct < 3; ++ct) {
#pragma unroll
    for (int r = 0; r < 8; ++r) {
      int m = row0 + r + half * 8;
      if (m < n) hw2[(size_t)m * 48 + ct * 16 + lrow] = acc[ct][r];
    }
  }
}

// ---------------- scatter: init with self-loop term, then edge atomics ----------------
// agg[i,:] = feat[i,:] * dinv[i]^2   (self-loop: norm = dinv[i]*dinv[i])

__global__ void k_scatter_init(const float* __restrict__ feat,
                               const float* __restrict__ dinv,
                               float* __restrict__ agg, int n, int stride4) {
  long long idx = (long long)blockIdx.x * blockDim.x + threadIdx.x;
  long long total = (long long)n * stride4;
  if (idx >= total) return;
  int node = (int)(idx / stride4);
  int c    = (int)(idx % stride4);
  float s = dinv[node]; s = s * s;
  const float4 v = *reinterpret_cast<const float4*>(feat + (size_t)node * stride4 * 4 + c * 4);
  float4 o; o.x = v.x * s; o.y = v.y * s; o.z = v.z * s; o.w = v.w * s;
  *reinterpret_cast<float4*>(agg + (size_t)node * stride4 * 4 + c * 4) = o;
}

// agg[dst,:] += feat[src,:] * dinv[src]*dinv[dst]   per edge

__global__ void k_scatter_edges(const long long* __restrict__ ei,
                                const float* __restrict__ dinv,
                                const float* __restrict__ feat,
                                float* __restrict__ agg, int E, int stride4) {
  long long idx = (long long)blockIdx.x * blockDim.x + threadIdx.x;
  long long total = (long long)E * stride4;
  if (idx >= total) return;
  int e = (int)(idx / stride4);
  int c = (int)(idx % stride4);
  int s = (int)ei[e];
  int d = (int)ei[(size_t)E + e];
  float nrm = dinv[s] * dinv[d];
  const float4 v = *reinterpret_cast<const float4*>(feat + (size_t)s * stride4 * 4 + c * 4);
  float* ap = agg + (size_t)d * stride4 * 4 + c * 4;
  unsafeAtomicAdd(ap + 0, v.x * nrm);
  unsafeAtomicAdd(ap + 1, v.y * nrm);
  unsafeAtomicAdd(ap + 2, v.z * nrm);
  unsafeAtomicAdd(ap + 3, v.w * nrm);
}

// ---------------- relu + bias (in place, layer1, 128 feats) ----------------

__global__ void k_relu_bias(float* __restrict__ agg, const float* __restrict__ b1, int n) {
  long long idx = (long long)blockIdx.x * blockDim.x + threadIdx.x;
  long long total = (long long)n * 32;
  if (idx >= total) return;
  int node = (int)(idx >> 5);
  int c    = (int)(idx & 31);
  float4 v = *reinterpret_cast<float4*>(agg + (size_t)node * 128 + c * 4);
  const float4 b = *reinterpret_cast<const float4*>(b1 + c * 4);
  v.x = fmaxf(v.x + b.x, 0.0f);
  v.y = fmaxf(v.y + b.y, 0.0f);
  v.z = fmaxf(v.z + b.z, 0.0f);
  v.w = fmaxf(v.w + b.w, 0.0f);
  *reinterpret_cast<float4*>(agg + (size_t)node * 128 + c * 4) = v;
}

// ---------------- bias + log_softmax over 40 classes ----------------

__global__ void k_logsoftmax(const float* __restrict__ agg2,
                             const float* __restrict__ b2,
                             float* __restrict__ out, int n) {
  int i = blockIdx.x * blockDim.x + threadIdx.x;
  if (i >= n) return;
  const float* row = agg2 + (size_t)i * 48;
  float v[40];
  float m = -3.4e38f;
#pragma unroll
  for (int c = 0; c < 40; ++c) { v[c] = row[c] + b2[c]; m = fmaxf(m, v[c]); }
  float s = 0.0f;
#pragma unroll
  for (int c = 0; c < 40; ++c) s += __expf(v[c] - m);
  float ls = __logf(s);
  float* orow = out + (size_t)i * 40;
#pragma unroll
  for (int c = 0; c < 40; ++c) orow[c] = v[c] - m - ls;
}

// ---------------- launch ----------------

extern "C" void kernel_launch(void* const* d_in, const int* in_sizes, int n_in,
                              void* d_out, int out_size, void* d_ws, size_t ws_size,
                              hipStream_t stream) {
  const float*     x  = (const float*)d_in[0];
  const long long* ei = (const long long*)d_in[1];
  const float*     W1 = (const float*)d_in[2];
  const float*     b1 = (const float*)d_in[3];
  const float*     W2 = (const float*)d_in[4];
  const float*     b2 = (const float*)d_in[5];
  float*           out = (float*)d_out;

  const int Nn = in_sizes[0] / 256;
  const int Ee = in_sizes[1] / 2;

  // workspace layout (floats)
  size_t Np = ((size_t)Nn + 255) & ~(size_t)255;
  float* ws   = (float*)d_ws;
  float* dinv = ws;                    // [Np]      (deg, then dinv in place)
  float* xw1  = dinv + Np;             // [Np*128]
  float* agg1 = xw1  + Np * 128;       // [Np*128]  (then h after relu)
  float* hw2  = agg1 + Np * 128;       // [Np*48]
  float* agg2 = hw2  + Np * 48;        // [Np*48]

  const int TB = 256;
  auto blocks = [](long long tot, int tb) { return (unsigned)((tot + tb - 1) / tb); };

  // norm
  k_deg_init <<<blocks(Nn, TB), TB, 0, stream>>>(dinv, Nn);
  k_deg_accum<<<blocks(Ee, TB), TB, 0, stream>>>(ei, dinv, Ee);
  k_dinv     <<<blocks(Nn, TB), TB, 0, stream>>>(dinv, Nn);

  // layer 1  (GEMM blocks cover 128 rows each)
  k_gemm1<<<blocks(((long long)Nn + 127) / 128, 1), 256, 0, stream>>>(x, W1, xw1, Nn);
  k_scatter_init <<<blocks((long long)Nn * 32, TB), TB, 0, stream>>>(xw1, dinv, agg1, Nn, 32);
  k_scatter_edges<<<blocks((long long)Ee * 32, TB), TB, 0, stream>>>(ei, dinv, xw1, agg1, Ee, 32);
  k_relu_bias    <<<blocks((long long)Nn * 32, TB), TB, 0, stream>>>(agg1, b1, Nn);

  // layer 2
  k_gemm2<<<blocks(((long long)Nn + 127) / 128, 1), 256, 0, stream>>>(agg1, W2, hw2, Nn);
  k_scatter_init <<<blocks((long long)Nn * 12, TB), TB, 0, stream>>>(hw2, dinv, agg2, Nn, 12);
  k_scatter_edges<<<blocks((long long)Ee * 12, TB), TB, 0, stream>>>(ei, dinv, hw2, agg2, Ee, 12);

  // output
  k_logsoftmax<<<blocks(Nn, TB), TB, 0, stream>>>(agg2, b2, out, Nn);
}
